// CrossAttention_747324309797
// MI455X (gfx1250) — matrix operand order repaired and checked by reading, not compiled, not run on hardware
//
#include <hip/hip_runtime.h>

typedef __attribute__((ext_vector_type(16))) __bf16 v16bf;
typedef __attribute__((ext_vector_type(8)))  float  v8f;

#define B_   4
#define H_   16
#define L_   2048
#define S_   2048
#define E_   64
#define DM_  1024
#define SCALE 0.125f
#define NT   (S_ / 32)

// LDS row strides in halves (16B-aligned rows, padded vs bank conflicts)
#define QS 72
#define KS 72
#define VS 40
#define PSD 40

union BFrag { unsigned int u[8]; uint4 q[2]; v16bf v; };

__device__ inline unsigned short f2bf(float x) {
  unsigned int u = __float_as_uint(x);
  u += 0x7FFFu + ((u >> 16) & 1u);      // round-to-nearest-even
  return (unsigned short)(u >> 16);
}

#if __has_builtin(__builtin_amdgcn_cvt_pk_bf16_f32)
typedef __attribute__((ext_vector_type(2))) __bf16 v2bf;
__device__ inline unsigned int f2bf2(float lo, float hi) {
  union { v2bf v; unsigned int u; } r;
  r.v = __builtin_amdgcn_cvt_pk_bf16_f32(lo, hi);
  return r.u;
}
#else
__device__ inline unsigned int f2bf2(float lo, float hi) {
  return (unsigned int)f2bf(lo) | ((unsigned int)f2bf(hi) << 16);
}
#endif

// DPP16 row_ror:N rotation within each 16-lane row (VALU, no DS traffic)
template <int CTRL>
__device__ inline float ror16(float x) {
  int r = __builtin_amdgcn_update_dpp(0, __float_as_int(x), CTRL, 0xF, 0xF, true);
  return __int_as_float(r);
}
__device__ inline float rowmax16(float x) {   // max over the 16-lane half
  x = fmaxf(x, ror16<0x121>(x));   // row_ror:1
  x = fmaxf(x, ror16<0x122>(x));   // row_ror:2
  x = fmaxf(x, ror16<0x124>(x));   // row_ror:4
  x = fmaxf(x, ror16<0x128>(x));   // row_ror:8
  return x;
}
__device__ inline float rowsum16(float x) {   // sum over the 16-lane half
  x += ror16<0x121>(x);
  x += ror16<0x122>(x);
  x += ror16<0x124>(x);
  x += ror16<0x128>(x);
  return x;
}

__device__ inline BFrag loadB(const unsigned short* p) {
  BFrag f;
  f.q[0] = *(const uint4*)(p);
  f.q[1] = *(const uint4*)(p + 16);
  return f;
}

__device__ inline v8f wmma_bf16(v16bf a, v16bf b, v8f c) {
  return __builtin_amdgcn_wmma_f32_16x16x32_bf16(false, a, false, b,
                                                 (short)0, c, false, false);
}

__global__ __launch_bounds__(128, 1)
void attn_fa_kernel(const float* __restrict__ Qg, const float* __restrict__ Kg,
                    const float* __restrict__ Vg, const float* __restrict__ Mg,
                    float* __restrict__ Og)
{
  __shared__ __align__(16) unsigned short Qs[64 * QS];
  __shared__ __align__(16) unsigned short Ks[2][32 * KS];
  __shared__ __align__(16) unsigned short Vt[2][64 * VS];
  __shared__ __align__(16) unsigned short Ps[4][16 * PSD];

  const int tid  = threadIdx.x;
  const int wv   = tid >> 5;
  const int lane = tid & 31;
  const int n    = lane & 15;
  const int hi   = lane >> 4;           // which 16-lane half

  const int mtile = blockIdx.x & 31;    // L/64 = 32
  const int bh    = blockIdx.x >> 5;
  const int b     = bh >> 4;
  const int h     = bh & 15;
  const int m0    = mtile * 64;

  const size_t kvbase = (size_t)(b) * S_ * DM_ + h * E_;

  // per-thread staging coordinates (4 float4s each for K and V tiles)
  int rowS[4], c4S[4];
  for (int i = 0; i < 4; ++i) {
    int idx4 = i * 128 + tid;           // 512 float4s per 32x64 tile
    rowS[i] = idx4 >> 4;                // s_local 0..31
    c4S[i]  = (idx4 & 15) * 4;          // e/d column
  }

  // ---- stage Q tile (64x64 fp32 -> bf16 LDS, row-major) ----
  for (int i = 0; i < 8; ++i) {
    int idx4 = i * 128 + tid;
    int row  = idx4 >> 4;
    int c4   = (idx4 & 15) * 4;
    float4 f = *(const float4*)(Qg + (size_t)(b * L_ + m0 + row) * DM_ + h * E_ + c4);
    *(uint2*)(&Qs[row * QS + c4]) = make_uint2(f2bf2(f.x, f.y), f2bf2(f.z, f.w));
  }

  // ---- stage K/V tile 0 into buffer 0 ----
  for (int i = 0; i < 4; ++i) {
    size_t gb = kvbase + (size_t)rowS[i] * DM_ + c4S[i];
    float4 fk = *(const float4*)(Kg + gb);
    *(uint2*)(&Ks[0][rowS[i] * KS + c4S[i]]) =
        make_uint2(f2bf2(fk.x, fk.y), f2bf2(fk.z, fk.w));
    float4 fv = *(const float4*)(Vg + gb);
    Vt[0][(c4S[i] + 0) * VS + rowS[i]] = f2bf(fv.x);
    Vt[0][(c4S[i] + 1) * VS + rowS[i]] = f2bf(fv.y);
    Vt[0][(c4S[i] + 2) * VS + rowS[i]] = f2bf(fv.z);
    Vt[0][(c4S[i] + 3) * VS + rowS[i]] = f2bf(fv.w);
  }
  __syncthreads();

  // ---- build Q A-fragments (16x32 bf16 layout, 2 E-steps) ----
  BFrag qf[2];
  {
    const int ka = hi ? 8 : 0;
    for (int f = 0; f < 2; ++f)
      for (int i = 0; i < 8; ++i) {
        int k0 = f * 32 + ((i < 4) ? (ka + 2 * i) : (16 + ka + 2 * (i - 4)));
        qf[f].u[i] = *(const unsigned int*)(&Qs[(wv * 16 + n) * QS + k0]);
      }
  }

  v8f acc[4];
  float rowmax[8], Zs[8];
  for (int c = 0; c < 4; ++c) acc[c] = (v8f){0,0,0,0,0,0,0,0};
  for (int r = 0; r < 8; ++r) { rowmax[r] = -1.0e30f; Zs[r] = 0.0f; }

  const int mrowb = m0 + wv * 16 + hi * 8;

  for (int t = 0; t < NT; ++t) {
    const int s0  = t * 32;
    const int cur = t & 1, nxt = cur ^ 1;
    const bool has_next = (t + 1 < NT);

    // ---- issue next tile's global loads into registers (overlap w/ compute) ----
    float4 fkr[4], fvr[4];
    if (has_next) {
      for (int i = 0; i < 4; ++i) {
        size_t gb = kvbase + (size_t)(s0 + 32 + rowS[i]) * DM_ + c4S[i];
        fkr[i] = *(const float4*)(Kg + gb);
        fvr[i] = *(const float4*)(Vg + gb);
      }
      if (t + 2 < NT) {                 // warm L2 two tiles ahead
        int idx = tid & 63, row = idx >> 1, hf = idx & 1;
        size_t gb = kvbase + (size_t)(s0 + 64 + row) * DM_ + hf * 32;
        const float* pp = (tid < 64) ? (Kg + gb) : (Vg + gb);
        __builtin_prefetch((const void*)pp, 0, 1);
      }
    }

    // ---- scores: 16x32 tile = 2 N-subtiles x 2 E-steps ----
    v8f c0 = (v8f){0,0,0,0,0,0,0,0};
    v8f c1 = (v8f){0,0,0,0,0,0,0,0};
    {
      const int kb = hi ? 8 : 0;
      BFrag k00 = loadB(&Ks[cur][n * KS + 0 + kb]);
      BFrag k01 = loadB(&Ks[cur][n * KS + 32 + kb]);
      BFrag k10 = loadB(&Ks[cur][(16 + n) * KS + 0 + kb]);
      BFrag k11 = loadB(&Ks[cur][(16 + n) * KS + 32 + kb]);
      c0 = wmma_bf16(qf[0].v, k00.v, c0);
      c0 = wmma_bf16(qf[1].v, k01.v, c0);
      c1 = wmma_bf16(qf[0].v, k10.v, c1);
      c1 = wmma_bf16(qf[1].v, k11.v, c1);
    }

    // ---- online softmax (Z without mask); reductions on VALU via DPP16 ----
    float p0[8], p1[8], al[8];
    for (int r = 0; r < 8; ++r) {
      float a = c0[r] * SCALE, bb = c1[r] * SCALE;
      float tm = rowmax16(fmaxf(a, bb));
      float nm = fmaxf(rowmax[r], tm);
      al[r] = __expf(rowmax[r] - nm);
      rowmax[r] = nm;
      p0[r] = __expf(a - nm);
      p1[r] = __expf(bb - nm);
      Zs[r] = Zs[r] * al[r] + rowsum16(p0[r] + p1[r]);
    }

    // ---- apply post-softmax mask, write P tile (C-layout -> LDS) ----
    for (int r = 0; r < 8; ++r) {
      float mk0 = Mg[(size_t)(mrowb + r) * S_ + s0 + n];
      float mk1 = Mg[(size_t)(mrowb + r) * S_ + s0 + 16 + n];
      Ps[wv][(hi * 8 + r) * PSD + n]      = f2bf(p0[r] * mk0);
      Ps[wv][(hi * 8 + r) * PSD + 16 + n] = f2bf(p1[r] * mk1);
    }
    asm volatile("" ::: "memory");   // same-wave LDS is in-order; stop reordering

    // ---- rescale accumulators ----
    for (int c = 0; c < 4; ++c)
      for (int r = 0; r < 8; ++r) acc[c][r] = acc[c][r] * al[r];

    // ---- P A-fragment, then PV WMMAs over 4 D-chunks ----
    BFrag pf;
    {
      const int ka = hi ? 8 : 0;
      for (int i = 0; i < 8; ++i) {
        int k0 = (i < 4) ? (ka + 2 * i) : (16 + ka + 2 * (i - 4));
        pf.u[i] = *(const unsigned int*)(&Ps[wv][n * PSD + k0]);
      }
    }
    {
      const int kb = hi ? 8 : 0;
      for (int c = 0; c < 4; ++c) {
        BFrag vf = loadB(&Vt[cur][(c * 16 + n) * VS + kb]);
        acc[c] = wmma_bf16(pf.v, vf.v, acc[c]);
      }
    }

    // ---- convert + store next tile into the other buffer ----
    if (has_next) {
      for (int i = 0; i < 4; ++i) {
        *(uint2*)(&Ks[nxt][rowS[i] * KS + c4S[i]]) =
            make_uint2(f2bf2(fkr[i].x, fkr[i].y), f2bf2(fkr[i].z, fkr[i].w));
        Vt[nxt][(c4S[i] + 0) * VS + rowS[i]] = f2bf(fvr[i].x);
        Vt[nxt][(c4S[i] + 1) * VS + rowS[i]] = f2bf(fvr[i].y);
        Vt[nxt][(c4S[i] + 2) * VS + rowS[i]] = f2bf(fvr[i].z);
        Vt[nxt][(c4S[i] + 3) * VS + rowS[i]] = f2bf(fvr[i].w);
      }
    }
    __syncthreads();   // one barrier per tile (double-buffered)
  }

  // ---- epilogue: divide by Z, store (B, L, H*D) ----
  for (int r = 0; r < 8; ++r) {
    float inv = 1.0f / Zs[r];
    for (int c = 0; c < 4; ++c)
      Og[(size_t)(b * L_ + mrowb + r) * DM_ + h * E_ + c * 16 + n] = acc[c][r] * inv;
  }
}

extern "C" void kernel_launch(void* const* d_in, const int* in_sizes, int n_in,
                              void* d_out, int out_size, void* d_ws, size_t ws_size,
                              hipStream_t stream) {
  const float* q = (const float*)d_in[0];
  const float* k = (const float*)d_in[1];
  const float* v = (const float*)d_in[2];
  const float* m = (const float*)d_in[3];
  float* o = (float*)d_out;
  (void)in_sizes; (void)n_in; (void)out_size; (void)d_ws; (void)ws_size;
  dim3 grid(B_ * H_ * (L_ / 64));   // 2048 blocks, one (b,h,64-row) tile each
  attn_fa_kernel<<<grid, 128, 0, stream>>>(q, k, v, m, o);
}
